// TargetMapRecovery_13168369729813
// MI455X (gfx1250) — matrix operand order repaired and checked by reading, not compiled, not run on hardware
//
#include <hip/hip_runtime.h>
#include <hip/hip_bf16.h>

#define NTHREADS 512
#define NWAVES   16
#define B_BATCH  64
#define NRX      8
#define NADC     256
#define THETA    181
#define RSAMP    256
#define MPAD     192                    // theta padded to 12 tiles of 16
#define NFLAT    (THETA * RSAMP)        // 46336
#define MASKW    ((MPAD * RSAMP) / 32)  // 1536 bitmask words
#define MAXATOMS 320
#define MAXIT    99                     // MAX_ITERATIONS - 1

// ---- CDNA5 WMMA types (wave32) ----
typedef __bf16 bf16_t;
typedef bf16_t v16bf __attribute__((ext_vector_type(16)));
typedef float  v8f   __attribute__((ext_vector_type(8)));

struct __attribute__((aligned(16))) U128 { unsigned x, y, z, w; };
union FragBF { v16bf v; U128 q[2]; };

__device__ __forceinline__ unsigned short f2bf(float f) {
  unsigned u = __float_as_uint(f);
  u += 0x7FFFu + ((u >> 16) & 1u);      // round-to-nearest-even
  return (unsigned short)(u >> 16);
}

__device__ __forceinline__ int getbit(const unsigned* m, int t, int r) {
  int f = (t << 8) | r;
  return (int)((m[f >> 5] >> (f & 31)) & 1u);
}

// ---- workspace layout: interleaved bf16 B operand, bdC[r][0:256]=re, [256:512]=im ----
#define WS_SCALE_OFF 0
#define WS_BDC_OFF   256
#define WS_TOTAL     (256 + RSAMP * 512 * 2)

// ---- dynamic LDS layout (bytes); imC[t][0:256]=re, [256:512]=im (bf16) ----
#define SM_IMC    0                            // u16[192*512]       196608
#define SM_AADJ   (SM_IMC + MPAD * 512 * 2)    // float2[181*8]       11584
#define SM_RESID  (SM_AADJ + THETA * NRX * 8)  // float2[8*256]       16384
#define SM_XDC    (SM_RESID + NRX * NADC * 8)  // float2[8*256]       16384
#define SM_XRE    (SM_XDC + NRX * RSAMP * 8)   // float[8*256]         8192
#define SM_MASK   (SM_XRE + NRX * NADC * 4)    // u32[1536]            6144
#define SM_GM     (SM_MASK + MASKW * 4)        // u32[1536]            6144
#define SM_FLIST  (SM_GM + MASKW * 4)          // int[320]             1280
#define SM_GLIST  (SM_FLIST + MAXATOMS * 4)    // float2[320]          2560
#define SM_REDV   (SM_GLIST + MAXATOMS * 8)    // float[NWAVES]          64
#define SM_REDI   (SM_REDV + NWAVES * 4)       // int[NWAVES]            64
#define SM_SF     (SM_REDI + NWAVES * 4)       // float[4]               16
#define SM_SI     (SM_SF + 16)                 // int[8]                 32
#define SMEM_BYTES (SM_SI + 32)                // ~265.5 KB < 320 KB/WGP

// ---------------- setup kernel 1: s = 1/sqrt(sum|x|) ----------------
__global__ void scale_kernel(const float* __restrict__ in, float* __restrict__ wsScale) {
  __shared__ double red[NTHREADS];
  double s = 0.0;
  for (int i = threadIdx.x; i < B_BATCH * NRX * NADC; i += NTHREADS)
    s += fabs((double)in[i]);
  red[threadIdx.x] = s;
  __syncthreads();
  for (int st = NTHREADS / 2; st > 0; st >>= 1) {
    if ((int)threadIdx.x < st) red[threadIdx.x] += red[threadIdx.x + st];
    __syncthreads();
  }
  if (threadIdx.x == 0) wsScale[0] = (float)(1.0 / sqrt(red[0]));
}

// ---------------- setup kernel 2: bf16 B operand, bdC[r][m]=re(conj(delay[m][r])), [256+m]=im ----
__global__ void bd_kernel(const double* __restrict__ dly,
                          unsigned short* __restrict__ bdC) {
  int i = blockIdx.x * blockDim.x + threadIdx.x;   // i = r*256 + m
  if (i >= NADC * RSAMP) return;
  int r = i >> 8, m = i & 255;
  double re = dly[2 * (m * RSAMP + r)];
  double im = dly[2 * (m * RSAMP + r) + 1];
  bdC[(r << 9) + m]       = f2bf((float)re);
  bdC[(r << 9) + 256 + m] = f2bf((float)(-im));
}

// ---------------- main kernel: one workgroup (16 waves) per batch ----------------
__global__ void __launch_bounds__(NTHREADS, 1)
TargetMapRecovery_13168369729813_kernel(const float* __restrict__ in,
                                        const double* __restrict__ angle,
                                        const double* __restrict__ dly,
                                        const float* __restrict__ wsScale,
                                        const unsigned short* __restrict__ bdC,
                                        float* __restrict__ out) {
  extern __shared__ unsigned char smem[];
  unsigned short* imC = (unsigned short*)(smem + SM_IMC);
  float2* aAdj   = (float2*)(smem + SM_AADJ);
  float2* resid  = (float2*)(smem + SM_RESID);
  float2* xDc    = (float2*)(smem + SM_XDC);
  float*  xRe    = (float*)(smem + SM_XRE);
  unsigned* mask = (unsigned*)(smem + SM_MASK);
  unsigned* gmm  = (unsigned*)(smem + SM_GM);
  int*    flist  = (int*)(smem + SM_FLIST);
  float2* gList  = (float2*)(smem + SM_GLIST);
  float*  redV   = (float*)(smem + SM_REDV);
  int*    redI   = (int*)(smem + SM_REDI);
  float*  sF     = (float*)(smem + SM_SF);   // [0]=damPrev
  int*    sI     = (int*)(smem + SM_SI);     // [0]=done [2]=selCount [4]=atomCnt

  const int tid  = threadIdx.x;
  const int lane = tid & 31;
  const int wave = tid >> 5;
  const int b    = blockIdx.x;
  const float s  = wsScale[0];

  // ---- init: A_adj = conj(angle)^T, x = s*input (imag 0), masks ----
  for (int e = tid; e < THETA * NRX; e += NTHREADS) {
    int t = e >> 3, n = e & 7;
    aAdj[e] = make_float2((float)angle[2 * (n * THETA + t)],
                          (float)(-angle[2 * (n * THETA + t) + 1]));
  }
  for (int e = tid; e < NRX * NADC; e += NTHREADS) {
    float xv = s * in[b * NRX * NADC + e];
    xRe[e] = xv;
    resid[e] = make_float2(xv, 0.f);
  }
  for (int e = tid; e < MASKW; e += NTHREADS) { mask[e] = 0u; gmm[e] = 0u; }
  if (tid == 0) { sF[0] = 0.f; sI[0] = 0; sI[2] = 0; }
  __syncthreads();

  // ---- xDc[n][r] = sum_m x[n][m] * conj(delay[m][r])  (for on-demand G0) ----
  for (int e = tid; e < NRX * RSAMP; e += NTHREADS) {
    int n = e >> 8, r = e & 255;
    float cr = 0.f, ci = 0.f;
    for (int m = 0; m < NADC; ++m) {
      float xv = xRe[n * NADC + m];
      const double* d = dly + 2 * (m * RSAMP + r);
      cr += xv * (float)d[0];
      ci -= xv * (float)d[1];
    }
    xDc[e] = make_float2(cr, ci);
  }
  __syncthreads();

  for (int iter = 1; iter <= MAXIT; ++iter) {
    // ---- Step A (K=8 VALU GEMM): interm = A_adj @ residual -> LDS bf16 (re/im interleaved rows) ----
    for (int e = tid; e < MPAD * RSAMP; e += NTHREADS) {
      int t = e >> 8, m = e & 255;
      float cr = 0.f, ci = 0.f;
      if (t < THETA) {
#pragma unroll
        for (int n = 0; n < NRX; ++n) {
          float2 a = aAdj[t * NRX + n];
          float2 x = resid[n * NADC + m];
          cr += a.x * x.x - a.y * x.y;
          ci += a.x * x.y + a.y * x.x;
        }
      }
      imC[(t << 9) + m]       = f2bf(cr);
      imC[(t << 9) + 256 + m] = f2bf(ci);
    }
    __syncthreads();

    // ---- Step B: dam = |interm @ Dc|^2 via bf16 WMMA; fused local argmax ----
    // A frags: ds_load_b128 with immediate offsets off one per-row base (re @ +0, im @ +512B)
    float bestV = -1.f; int bestI = 0;
    const int k0a = (lane >> 4) << 3;   // A frag: K {0..7,16..23} / {8..15,24..31}
  const int k0b = (lane >> 4) << 3;   // B frag: K {0..7,16..23} / {8..15,24..31}
    for (int tile = wave; tile < (MPAD / 16) * (RSAMP / 16); tile += NWAVES) {
      const int mt = tile >> 4, nt = tile & 15;   // nt constant per wave -> B stays in VGPRs
      const unsigned short* aRow = imC + (((mt << 4) + (lane & 15)) << 9);
      const unsigned short* bRow = bdC + (((nt << 4) + (lane & 15)) << 9);
      v8f accRR = {0.f,0.f,0.f,0.f,0.f,0.f,0.f,0.f};
      v8f accII = accRR, accRI = accRR, accIR = accRR;
      for (int k = 0; k < NADC; k += 32) {
        FragBF aRe, aIm, bRe, bIm;
        aRe.q[0] = *(const U128*)(aRow + k + k0a);
        aRe.q[1] = *(const U128*)(aRow + k + k0a + 16);
        aIm.q[0] = *(const U128*)(aRow + 256 + k + k0a);
        aIm.q[1] = *(const U128*)(aRow + 256 + k + k0a + 16);
        bRe.q[0] = *(const U128*)(bRow + k + k0b);
        bRe.q[1] = *(const U128*)(bRow + k + k0b + 16);
        bIm.q[0] = *(const U128*)(bRow + 256 + k + k0b);
        bIm.q[1] = *(const U128*)(bRow + 256 + k + k0b + 16);
        accRR = __builtin_amdgcn_wmma_f32_16x16x32_bf16(false, aRe.v, false, bRe.v, (short)0, accRR, false, false);
        accII = __builtin_amdgcn_wmma_f32_16x16x32_bf16(false, aIm.v, false, bIm.v, (short)0, accII, false, false);
        accRI = __builtin_amdgcn_wmma_f32_16x16x32_bf16(false, aRe.v, false, bIm.v, (short)0, accRI, false, false);
        accIR = __builtin_amdgcn_wmma_f32_16x16x32_bf16(false, aIm.v, false, bRe.v, (short)0, accIR, false, false);
      }
      const int tBase = (mt << 4) + ((lane >> 4) << 3);
      const int r = (nt << 4) + (lane & 15);
#pragma unroll
      for (int j = 0; j < 8; ++j) {
        int t = tBase + j;
        float re = accRR[j] - accII[j];
        float im = accRI[j] + accIR[j];
        float p = re * re + im * im;
        int flat = (t << 8) | r;
        if (t < THETA && (p > bestV || (p == bestV && flat < bestI))) { bestV = p; bestI = flat; }
      }
    }
    for (int off = 16; off; off >>= 1) {       // wave32 argmax, tie -> lowest idx
      float ov = __shfl_xor(bestV, off, 32);
      int   oi = __shfl_xor(bestI, off, 32);
      if (ov > bestV || (ov == bestV && oi < bestI)) { bestV = ov; bestI = oi; }
    }
    if (lane == 0) { redV[wave] = bestV; redI[wave] = bestI; }
    __syncthreads();
    if (tid == 0) {
      float v = redV[0]; int idx = redI[0];
      for (int q = 1; q < NWAVES; ++q)
        if (redV[q] > v || (redV[q] == v && redI[q] < idx)) { v = redV[q]; idx = redI[q]; }
      sI[0] += (v > 3.f * sF[0]) ? 1 : 0;      // done_new
      sF[0] = v;                               // dam_prev = dam_max
      unsigned wd = (unsigned)idx >> 5, bm = 1u << (idx & 31);
      if (!(mask[wd] & bm)) { mask[wd] |= bm; sI[2] += 1; }
    }
    __syncthreads();

    // ---- sobel (summed dy+dx kernel, reflect pad) masked argmax ----
    float sbV = -1.f; int sbI = 0;
    for (int e = tid; e < NFLAT; e += NTHREADS) {
      int t = e >> 8, r = e & 255;
      float score = 0.f;
      if (!((mask[e >> 5] >> (e & 31)) & 1u)) {
        int tm = (t == 0) ? 1 : t - 1;
        int tp = (t == THETA - 1) ? THETA - 2 : t + 1;
        int rm = (r == 0) ? 1 : r - 1;
        int rp = (r == RSAMP - 1) ? RSAMP - 2 : r + 1;
        int ss = -2 * getbit(mask, tm, rm) - 2 * getbit(mask, tm, r)
                 - 2 * getbit(mask, t, rm)  + 2 * getbit(mask, t, rp)
                 + 2 * getbit(mask, tp, r)  + 2 * getbit(mask, tp, rp);
        score = (float)(ss < 0 ? -ss : ss);
      }
      if (score > sbV || (score == sbV && e < sbI)) { sbV = score; sbI = e; }
    }
    for (int off = 16; off; off >>= 1) {
      float ov = __shfl_xor(sbV, off, 32);
      int   oi = __shfl_xor(sbI, off, 32);
      if (ov > sbV || (ov == sbV && oi < sbI)) { sbV = ov; sbI = oi; }
    }
    if (lane == 0) { redV[wave] = sbV; redI[wave] = sbI; }
    __syncthreads();
    if (tid == 0) {
      float v = redV[0]; int idx = redI[0];
      for (int q = 1; q < NWAVES; ++q)
        if (redV[q] > v || (redV[q] == v && redI[q] < idx)) { v = redV[q]; idx = redI[q]; }
      if (iter != sI[2]) {                     // eq = (i == sum(upd)); add sobel pixel if !eq
        unsigned wd = (unsigned)idx >> 5, bm = 1u << (idx & 31);
        if (!(mask[wd] & bm)) { mask[wd] |= bm; sI[2] += 1; }
      }
    }
    __syncthreads();

    if (sI[0] == 0)                            // gm frozen once done
      for (int e = tid; e < MASKW; e += NTHREADS) gmm[e] = mask[e];
    __syncthreads();
    if (sI[0] > 0) break;                      // gm frozen -> output fixed -> exit

    // ---- sparse recon: residual = x - angle @ (mask*G0) @ delay^T ----
    if (tid == 0) {                            // deterministic atom list
      int c = 0;
      for (int wd = 0; wd < MASKW && c < MAXATOMS; ++wd) {
        unsigned v = mask[wd];
        while (v && c < MAXATOMS) {
          int bp = __ffs((int)v) - 1;
          v &= v - 1;
          flist[c++] = (wd << 5) | bp;
        }
      }
      sI[4] = c;
    }
    __syncthreads();
    const int cnt = sI[4];
    for (int j = tid; j < cnt; j += NTHREADS) {     // G0 on demand (8 cmuls)
      int flat = flist[j], t = flat >> 8, r = flat & 255;
      float gr = 0.f, gi = 0.f;
#pragma unroll
      for (int n = 0; n < NRX; ++n) {
        float2 a = aAdj[t * NRX + n];
        float2 xd = xDc[n * RSAMP + r];
        gr += a.x * xd.x - a.y * xd.y;
        gi += a.x * xd.y + a.y * xd.x;
      }
      gList[j] = make_float2(gr * (1.f / 2048.f), gi * (1.f / 2048.f));
    }
    __syncthreads();
    for (int e = tid; e < NRX * NADC; e += NTHREADS) {
      int n = e >> 8, m = e & 255;
      float accR = 0.f, accI = 0.f;
      for (int j = 0; j < cnt; ++j) {
        int flat = flist[j], t = flat >> 8, r = flat & 255;
        float2 g = gList[j];
        float2 aa = aAdj[t * NRX + n];             // angle[n][t] = conj(aAdj)
        float c1r = aa.x * g.x + aa.y * g.y;
        float c1i = aa.x * g.y - aa.y * g.x;
        const double* d = dly + 2 * (m * RSAMP + r);
        float dr = (float)d[0], di = (float)d[1];
        accR += c1r * dr - c1i * di;
        accI += c1r * di + c1i * dr;
      }
      resid[e] = make_float2(xRe[e] - accR, -accI);
    }
    __syncthreads();
  }

  // ---- emit gm (flat order t*256+r == reference reshape) ----
  for (int e = tid; e < NFLAT; e += NTHREADS)
    out[b * NFLAT + e] = ((gmm[e >> 5] >> (e & 31)) & 1u) ? 1.0f : 0.0f;
}

extern "C" void kernel_launch(void* const* d_in, const int* in_sizes, int n_in,
                              void* d_out, int out_size, void* d_ws, size_t ws_size,
                              hipStream_t stream) {
  (void)in_sizes; (void)n_in; (void)out_size; (void)ws_size;
  const float*  in    = (const float*)d_in[0];
  const double* angle = (const double*)d_in[1];   // complex128 pairs (8,181)
  const double* dly   = (const double*)d_in[2];   // complex128 pairs (256,256)
  float* out = (float*)d_out;

  unsigned char* ws = (unsigned char*)d_ws;
  float*          wsScale = (float*)(ws + WS_SCALE_OFF);
  unsigned short* bdC     = (unsigned short*)(ws + WS_BDC_OFF);

  (void)hipFuncSetAttribute(
      reinterpret_cast<const void*>(&TargetMapRecovery_13168369729813_kernel),
      hipFuncAttributeMaxDynamicSharedMemorySize, SMEM_BYTES);

  scale_kernel<<<1, NTHREADS, 0, stream>>>(in, wsScale);
  bd_kernel<<<(NADC * RSAMP + 255) / 256, 256, 0, stream>>>(dly, bdC);
  TargetMapRecovery_13168369729813_kernel<<<B_BATCH, NTHREADS, SMEM_BYTES, stream>>>(
      in, angle, dly, wsScale, bdC, out);
}
